// DrugGraphNet_4827543241416
// MI455X (gfx1250) — matrix operand order repaired and checked by reading, not compiled, not run on hardware
//
#include <hip/hip_runtime.h>
#include <hip/hip_bf16.h>

// ---------------- problem constants (from reference) ----------------
#define N_NODES_C      245760
#define N_GRAPHS_C     8192
#define N_EDGES_C      1000000
#define NODES_PER_G    30
#define F_NODE_C       78
#define F_CELL_C       1000

typedef __attribute__((ext_vector_type(16))) __bf16 v16bf;
typedef __attribute__((ext_vector_type(8)))  float  v8f;

__device__ __forceinline__ __bf16 f2bf(float f) {
  unsigned u = __builtin_bit_cast(unsigned, f);
  u += 0x7FFFu + ((u >> 16) & 1u);              // round-to-nearest-even
  unsigned short h = (unsigned short)(u >> 16);
  return __builtin_bit_cast(__bf16, h);
}

// ---------------------------------------------------------------
// GEMM: C[M,N] = act( A[M,K](f32,row-major,lda) @ W[K,N](f32,row-major,ldw) + bias )
// One wave32 computes a 16 x (16*NT) tile: NT accumulators share one A
// fragment per K-step, cutting redundant A reads by NT (A is the dominant
// HBM term; B is L2-resident). REQUIRES M % 16 == 0, N % (16*NT) == 0.
// K arbitrary: unpredicated main loop over K&~31 plus one branch-free
// masked tail (K=78, K=1000).
// ---------------------------------------------------------------
template <int NT>
__global__ __launch_bounds__(32)
void gemm_wmma_bf16(const float* __restrict__ A, int lda,
                    const float* __restrict__ W, int ldw,
                    const float* __restrict__ bias,
                    float* __restrict__ C, int ldc, int coloff,
                    int K, int doRelu) {
  const int lane = threadIdx.x;        // 0..31
  const int half = lane >> 4;          // 0 | 1
  const int l15  = lane & 15;

  const long long rowA    = (long long)blockIdx.x * 16 + l15;
  const int       colBase = blockIdx.y * (16 * NT);
  const int kbA = half ? 8  : 0;       // A-frag K offset for upper lane half
  const int kbB = half ? 8  : 0;       // B-frag K offset for upper lane half

  const float* __restrict__ aPtr = A + rowA * lda;

  v8f acc[NT];
#pragma unroll
  for (int t = 0; t < NT; ++t) acc[t] = (v8f){};

  const int kFull = K & ~31;

  // ---- main loop: zero predication ----
  for (int k0 = 0; k0 < kFull; k0 += 32) {
    v16bf a;
#pragma unroll
    for (int i = 0; i < 8; ++i) {
      // A 16x32 bf16 layout: elems 0..7 -> K=kbA+0..7 ; elems 8..15 -> K=kbA+16..23
      a[i]     = f2bf(aPtr[k0 + kbA + i]);
      a[i + 8] = f2bf(aPtr[k0 + kbA + 16 + i]);
    }
#pragma unroll
    for (int t = 0; t < NT; ++t) {
      const float* __restrict__ bPtr = W + colBase + t * 16 + l15;
      v16bf b;
#pragma unroll
      for (int i = 0; i < 16; ++i) {
        // B 32x16 bf16 layout: elems 0..7 -> K=kbB+0..7 ; elems 8..15 -> K=kbB+16..23
        b[i] = f2bf(bPtr[(long long)(k0 + kbB + i + ((i >= 8) ? 8 : 0)) * ldw]);
      }
      acc[t] = __builtin_amdgcn_wmma_f32_16x16x32_bf16(
          false, a, false, b, (short)0, acc[t], false, false);
    }
  }

  // ---- branch-free masked tail (only when K % 32 != 0) ----
  if (kFull < K) {
    v16bf a;
#pragma unroll
    for (int i = 0; i < 16; ++i) {
      int ka  = kFull + kbA + i + ((i >= 8) ? 8 : 0);
      int kac = (ka < K) ? ka : (K - 1);              // clamped, always in bounds
      a[i] = f2bf(aPtr[kac] * ((ka < K) ? 1.0f : 0.0f));
    }
#pragma unroll
    for (int t = 0; t < NT; ++t) {
      const float* __restrict__ bPtr = W + colBase + t * 16 + l15;
      v16bf b;
#pragma unroll
      for (int i = 0; i < 16; ++i) {
        int kb  = kFull + kbB + i + ((i >= 8) ? 8 : 0);
        int kbc = (kb < K) ? kb : (K - 1);
        b[i] = f2bf(bPtr[(long long)kbc * ldw] * ((kb < K) ? 1.0f : 0.0f));
      }
      acc[t] = __builtin_amdgcn_wmma_f32_16x16x32_bf16(
          false, a, false, b, (short)0, acc[t], false, false);
    }
  }

  // ---- epilogue: no guards needed (M,N multiples of tile) ----
#pragma unroll
  for (int t = 0; t < NT; ++t) {
    int col  = colBase + t * 16 + l15;
    float bb = bias ? bias[col] : 0.0f;
    float* __restrict__ cPtr =
        C + ((long long)blockIdx.x * 16 + (half ? 8 : 0)) * ldc + coloff + col;
#pragma unroll
    for (int r = 0; r < 8; ++r) {
      float v = acc[t][r] + bb;
      if (doRelu) v = fmaxf(v, 0.0f);
      cPtr[(long long)r * ldc] = v;   // C/D layout: lanes 16-31 hold M=8..15
    }
  }
}

// ---------------- degree / norm ----------------
__global__ void deg_init(float* deg, int n) {
  int i = blockIdx.x * blockDim.x + threadIdx.x;
  if (i < n) deg[i] = 1.0f;                 // self-loop contribution
}
__global__ void deg_accum(const int* __restrict__ dst, float* deg, int nE) {
  int e = blockIdx.x * blockDim.x + threadIdx.x;
  if (e < nE) atomicAdd(&deg[dst[e]], 1.0f);
}
__global__ void rsqrt_inplace(float* deg, int n) {
  int i = blockIdx.x * blockDim.x + threadIdx.x;
  if (i < n) deg[i] = rsqrtf(deg[i]);
}

// ---------------- normalized scatter-add (edges + self loops) ----------------
__global__ void scatter_edges(const int* __restrict__ ei,     // [2, N_EDGES] flat
                              const float* __restrict__ dinv,
                              const float* __restrict__ t,    // [N, F] linear output
                              float* __restrict__ agg,        // [N, F] pre-zeroed
                              int F) {
  const long long total  = (long long)(N_EDGES_C + N_NODES_C) * F;
  const long long stride = (long long)gridDim.x * blockDim.x;
  for (long long idx = (long long)blockIdx.x * blockDim.x + threadIdx.x;
       idx < total; idx += stride) {
    int e = (int)(idx / F);
    int f = (int)(idx - (long long)e * F);
    int src, dst;
    if (e < N_EDGES_C) { src = ei[e]; dst = ei[N_EDGES_C + e]; }
    else               { src = dst = e - N_EDGES_C; }
    float w = dinv[src] * dinv[dst];
    atomicAdd(&agg[(long long)dst * F + f], t[(long long)src * F + f] * w);
  }
}

// ---------------- bias + relu (post-aggregation) ----------------
__global__ void bias_relu(float* __restrict__ h, const float* __restrict__ b,
                          int F, long long total) {
  long long idx = (long long)blockIdx.x * blockDim.x + threadIdx.x;
  if (idx < total) {
    int f = (int)(idx % F);
    h[idx] = fmaxf(h[idx] + b[f], 0.0f);
  }
}

// ---------------- mean pool over 30 nodes / graph ----------------
__global__ void pool_mean(const float* __restrict__ h, float* __restrict__ mean, int F) {
  long long idx = (long long)blockIdx.x * blockDim.x + threadIdx.x;
  long long total = (long long)N_GRAPHS_C * F;
  if (idx >= total) return;
  int g = (int)(idx / F), f = (int)(idx - (long long)g * F);
  float s = 0.0f;
  for (int j = 0; j < NODES_PER_G; ++j)
    s += h[((long long)(g * NODES_PER_G + j)) * F + f];
  mean[idx] = s * (1.0f / NODES_PER_G);
}

// ---------------- final 32 -> 1 ----------------
__global__ void final_fc(const float* __restrict__ z, const float* __restrict__ Wo,
                         const float* __restrict__ bo, float* __restrict__ out) {
  int g = blockIdx.x * blockDim.x + threadIdx.x;
  if (g >= N_GRAPHS_C) return;
  float s = bo[0];
  for (int j = 0; j < 32; ++j) s += z[g * 32 + j] * Wo[j];
  out[g] = s;
}

// ---------------- host launch ----------------
static inline int capBlocks(long long total, int tpb) {
  long long b = (total + tpb - 1) / tpb;
  if (b > (1LL << 20)) b = (1LL << 20);
  return (int)b;
}

extern "C" void kernel_launch(void* const* d_in, const int* in_sizes, int n_in,
                              void* d_out, int out_size, void* d_ws, size_t ws_size,
                              hipStream_t stream) {
  (void)in_sizes; (void)n_in; (void)out_size; (void)ws_size;
  const float* x    = (const float*)d_in[0];
  const int*   ei   = (const int*)  d_in[1];   // [2, N_EDGES]
  /* batch (d_in[2]) is implicit: node i -> graph i/30 */
  const float* cell = (const float*)d_in[3];
  const float* W1 = (const float*)d_in[4],  *b1 = (const float*)d_in[5];
  const float* W2 = (const float*)d_in[6],  *b2 = (const float*)d_in[7];
  const float* W3 = (const float*)d_in[8],  *b3 = (const float*)d_in[9];
  const float* Wd = (const float*)d_in[10], *bd = (const float*)d_in[11];
  const float* Wc1= (const float*)d_in[12], *bc1= (const float*)d_in[13];
  const float* Wc2= (const float*)d_in[14], *bc2= (const float*)d_in[15];
  const float* Wm1= (const float*)d_in[16], *bm1= (const float*)d_in[17];
  const float* Wm2= (const float*)d_in[18], *bm2= (const float*)d_in[19];
  const float* Wo = (const float*)d_in[20], *bo = (const float*)d_in[21];
  float* out = (float*)d_out;

  // workspace layout (floats)
  float* ws = (float*)d_ws;
  float* dinv = ws;                          ws += N_NODES_C;
  float* bufA = ws;                          ws += (long long)N_NODES_C * 256; // linear out
  float* bufB = ws;                          ws += (long long)N_NODES_C * 256; // agg / h
  float* meanB= ws;                          ws += (long long)N_GRAPHS_C * 256;
  float* cbuf = ws;                          ws += (long long)N_GRAPHS_C * 128;
  float* comb = ws;                          ws += (long long)N_GRAPHS_C * 128; // [drug|cell]
  float* z1   = ws;                          ws += (long long)N_GRAPHS_C * 64;
  float* z2   = ws;                          /* N_GRAPHS*32 */

  const int TPB = 256;
  // 1) degrees -> dinv
  deg_init <<<(N_NODES_C + TPB - 1) / TPB, TPB, 0, stream>>>(dinv, N_NODES_C);
  deg_accum<<<(N_EDGES_C + TPB - 1) / TPB, TPB, 0, stream>>>(ei + N_EDGES_C, dinv, N_EDGES_C);
  rsqrt_inplace<<<(N_NODES_C + TPB - 1) / TPB, TPB, 0, stream>>>(dinv, N_NODES_C);

  const int mT = N_NODES_C / 16;   // 15360
  const int gT = N_GRAPHS_C / 16;  // 512

  // ---- GCN layer 1: 78 -> 64  (NT=4 covers all 64 cols; A read once) ----
  gemm_wmma_bf16<4><<<dim3(mT, 1), 32, 0, stream>>>(x, F_NODE_C, W1, 64, nullptr,
                                                    bufA, 64, 0, F_NODE_C, 0);
  hipMemsetAsync(bufB, 0, (size_t)N_NODES_C * 64 * sizeof(float), stream);
  {
    long long tot = (long long)(N_EDGES_C + N_NODES_C) * 64;
    scatter_edges<<<capBlocks(tot, TPB), TPB, 0, stream>>>(ei, dinv, bufA, bufB, 64);
    bias_relu<<<capBlocks((long long)N_NODES_C * 64, TPB), TPB, 0, stream>>>(
        bufB, b1, 64, (long long)N_NODES_C * 64);
  }
  // ---- GCN layer 2: 64 -> 128  (NT=8 covers all 128 cols) ----
  gemm_wmma_bf16<8><<<dim3(mT, 1), 32, 0, stream>>>(bufB, 64, W2, 128, nullptr,
                                                    bufA, 128, 0, 64, 0);
  hipMemsetAsync(bufB, 0, (size_t)N_NODES_C * 128 * sizeof(float), stream);
  {
    long long tot = (long long)(N_EDGES_C + N_NODES_C) * 128;
    scatter_edges<<<capBlocks(tot, TPB), TPB, 0, stream>>>(ei, dinv, bufA, bufB, 128);
    bias_relu<<<capBlocks((long long)N_NODES_C * 128, TPB), TPB, 0, stream>>>(
        bufB, b2, 128, (long long)N_NODES_C * 128);
  }
  // ---- GCN layer 3: 128 -> 256  (NT=8, 2 column groups; A read twice) ----
  gemm_wmma_bf16<8><<<dim3(mT, 2), 32, 0, stream>>>(bufB, 128, W3, 256, nullptr,
                                                    bufA, 256, 0, 128, 0);
  hipMemsetAsync(bufB, 0, (size_t)N_NODES_C * 256 * sizeof(float), stream);
  {
    long long tot = (long long)(N_EDGES_C + N_NODES_C) * 256;
    scatter_edges<<<capBlocks(tot, TPB), TPB, 0, stream>>>(ei, dinv, bufA, bufB, 256);
    bias_relu<<<capBlocks((long long)N_NODES_C * 256, TPB), TPB, 0, stream>>>(
        bufB, b3, 256, (long long)N_NODES_C * 256);
  }
  // ---- mean pool ----
  pool_mean<<<capBlocks((long long)N_GRAPHS_C * 256, TPB), TPB, 0, stream>>>(bufB, meanB, 256);

  // ---- drug embed: [8192,256] @ Wd -> comb[:, 0:64] ----
  gemm_wmma_bf16<4><<<dim3(gT, 1), 32, 0, stream>>>(meanB, 256, Wd, 64, bd,
                                                    comb, 128, 0, 256, 0);
  // ---- cell MLP: relu(cell @ Wc1 + bc1) @ Wc2 + bc2 -> comb[:, 64:128] ----
  gemm_wmma_bf16<8><<<dim3(gT, 1), 32, 0, stream>>>(cell, F_CELL_C, Wc1, 128, bc1,
                                                    cbuf, 128, 0, F_CELL_C, 1);
  gemm_wmma_bf16<4><<<dim3(gT, 1), 32, 0, stream>>>(cbuf, 128, Wc2, 64, bc2,
                                                    comb, 128, 64, 128, 0);
  // ---- combiner ----
  gemm_wmma_bf16<4><<<dim3(gT, 1), 32, 0, stream>>>(comb, 128, Wm1, 64, bm1,
                                                    z1, 64, 0, 128, 1);
  gemm_wmma_bf16<2><<<dim3(gT, 1), 32, 0, stream>>>(z1, 64, Wm2, 32, bm2,
                                                    z2, 32, 0, 64, 1);
  final_fc<<<(N_GRAPHS_C + TPB - 1) / TPB, TPB, 0, stream>>>(z2, Wo, bo, out);
}
